// TMTM_40209483825627
// MI455X (gfx1250) — matrix operand order repaired and checked by reading, not compiled, not run on hardware
//
#include <hip/hip_runtime.h>

#define H 128
#define RNUM 12
#define FD 1582
#define TILE 64
#define S1K 1632   // stage1 padded K: cat[0,32) num[32,96) des[96,864) tweet[864,1632)

typedef __attribute__((ext_vector_type(16))) __bf16 v16bf;
typedef __attribute__((ext_vector_type(8)))  float  v8f;

static __device__ __forceinline__ __bf16 f2bf(float f) {
  union { float f; unsigned int u; } in; in.f = f;
  unsigned int u = in.u;
  unsigned int r = u + 0x7FFFu + ((u >> 16) & 1u);   // round-to-nearest-even
  union { unsigned short s; __bf16 b; } out; out.s = (unsigned short)(r >> 16);
  return out.b;
}

static __device__ __forceinline__ float lrelu(float v) { return v > 0.f ? v : 0.01f * v; }

// CDNA5 async global->LDS copy, 16B per lane (GLOBAL_LOAD_ASYNC_TO_LDS_B128, ASYNCcnt).
static __device__ __forceinline__ void async_cp16(void* lds, const void* gaddr) {
  unsigned loff = (unsigned)(unsigned long long)lds;   // low 32 bits = LDS offset
  asm volatile("global_load_async_to_lds_b128 %0, %1, off"
               :: "v"(loff), "v"(gaddr) : "memory");
}
static __device__ __forceinline__ void async_wait0() {
  asm volatile("s_wait_asynccnt 0x0" ::: "memory");
}

// ---------------------------------------------------------------- edge prep
__global__ void count_kernel(const int* __restrict__ dst, const int* __restrict__ et,
                             float* __restrict__ cnt, int E, int Nn) {
  int e = blockIdx.x * blockDim.x + threadIdx.x;
  if (e < E) atomicAdd(&cnt[(long)et[e] * Nn + dst[e]], 1.0f);
}

__global__ void hist_kernel(const int* __restrict__ et, int* __restrict__ hist, int E) {
  int e = blockIdx.x * blockDim.x + threadIdx.x;
  if (e < E) atomicAdd(&hist[et[e]], 1);
}

__global__ void scan_kernel(const int* __restrict__ hist, int* __restrict__ cursor) {
  if (threadIdx.x == 0 && blockIdx.x == 0) {
    int a = 0;
    for (int r = 0; r < RNUM; ++r) {
      cursor[r] = a;
      a += ((hist[r] + 63) >> 6) << 6;   // 64-align each relation segment
    }
  }
}

__global__ void scatter_kernel(const int* __restrict__ src, const int* __restrict__ dst,
                               const int* __restrict__ et, const float* __restrict__ cnt,
                               int* __restrict__ cursor,
                               int* __restrict__ ssrc, int* __restrict__ sdst,
                               int* __restrict__ srel, float* __restrict__ snrm,
                               int E, int Nn) {
  int e = blockIdx.x * blockDim.x + threadIdx.x;
  if (e >= E) return;
  int r = et[e];
  int p = atomicAdd(&cursor[r], 1);
  int d = dst[e];
  ssrc[p] = src[e];
  sdst[p] = d;
  srel[p] = r;
  float c = cnt[(long)r * Nn + d];
  snrm[p] = 1.0f / fmaxf(c, 1.0f);
}

// ---------------------------------------------------------------- f32 -> bf16 convert
__global__ void fconv_kernel(const float* __restrict__ s, __bf16* __restrict__ d, long n) {
  long i = (long)blockIdx.x * blockDim.x + threadIdx.x;
  if (i < n) d[i] = f2bf(s[i]);
}

// ---------------------------------------------------------------- input embed (WMMA)
// 16 nodes/block; feature row remapped into zero-padded segment-major K space.
// Output col tiles: t0,t1=des  t2,t3=tweet  t4,t5=num  t6,t7=cat (matches concat order).
__global__ void stage1_kernel(const float* __restrict__ feat,
                              const float* __restrict__ Wd, const float* __restrict__ bd,
                              const float* __restrict__ Wt, const float* __restrict__ bt,
                              const float* __restrict__ Wn, const float* __restrict__ bn,
                              const float* __restrict__ Wc, const float* __restrict__ bc,
                              __bf16* __restrict__ xb, int Nn) {
  __shared__ __bf16 ft[16][S1K + 8];
  __shared__ __bf16 ws[S1K][32];
  __shared__ float  bs[128];
  int tid = threadIdx.x;
  int nodeBase = blockIdx.x * 16;
  __bf16 z = f2bf(0.f);
  for (int i = tid; i < 16 * (S1K + 8); i += 256) (&ft[0][0])[i] = z;
  for (int i = tid; i < S1K * 32; i += 256)       (&ws[0][0])[i] = z;
  if (tid < 128) {
    int s = tid >> 5, c = tid & 31;
    bs[tid] = (s == 0 ? bd : s == 1 ? bt : s == 2 ? bn : bc)[c];
  }
  __syncthreads();
  for (int i = tid; i < 768 * 32; i += 256) ws[96  + (i >> 5)][i & 31] = f2bf(Wd[i]);
  for (int i = tid; i < 768 * 32; i += 256) ws[864 + (i >> 5)][i & 31] = f2bf(Wt[i]);
  for (int i = tid; i < 34 * 32;  i += 256) ws[32  + (i >> 5)][i & 31] = f2bf(Wn[i]);
  for (int i = tid; i < 12 * 32;  i += 256) ws[0   + (i >> 5)][i & 31] = f2bf(Wc[i]);
  for (int i = tid; i < 16 * FD; i += 256) {
    int r = i / FD, c = i % FD;
    int row = nodeBase + r;
    float v = (row < Nn) ? feat[(size_t)row * FD + c] : 0.f;
    int mc;
    if      (c < 12)  mc = c;               // cat
    else if (c < 46)  mc = 32 + (c - 12);   // num
    else if (c < 814) mc = 96 + (c - 46);   // des
    else              mc = 864 + (c - 814); // tweet
    ft[r][mc] = f2bf(v);
  }
  __syncthreads();
  int lane = tid & 31;
  int t = tid >> 5;                    // col tile, wave-uniform
  int seg = t >> 1;
  int kbase  = (seg == 0) ? 96 : (seg == 1) ? 864 : (seg == 2) ? 32 : 0;
  int ksteps = (seg <= 1) ? 24 : (seg == 2) ? 2 : 1;
  int kbA = (lane & 16) ? 8 : 0;
  int m   = lane & 15;
  int kB  = (lane & 15) + ((lane & 16) ? 16 : 0);
  int nb  = (t & 1) * 16;
  v8f acc = {};
  for (int kk = 0; kk < ksteps; ++kk) {
    v16bf a, b;
    #pragma unroll
    for (int i = 0; i < 16; ++i) {
      int k = ((i >> 3) << 4) + kbA + (i & 7);
      a[i] = ft[m][kbase + kk * 32 + k];
      b[i] = ws[kbase + kk * 32 + k][nb + m];
    }
    acc = __builtin_amdgcn_wmma_f32_16x16x32_bf16(false, a, false, b,
                                                  (short)0, acc, false, false);
  }
  int mo  = (lane & 16) ? 8 : 0;
  int col = t * 16 + (lane & 15);
  #pragma unroll
  for (int j = 0; j < 8; ++j) {
    int row = nodeBase + j + mo;
    if (row < Nn) xb[(size_t)row * H + col] = f2bf(lrelu(acc[j] + bs[col]));
  }
}

// ---------------------------------------------------------------- dense WMMA GEMM
// C = op(A[M,128] @ W[128,128] + bias). A/W are bf16 in global, staged to LDS with
// async-tensor copies. Optional f32 and/or bf16 outputs.
__global__ void gemm128_kernel(const __bf16* __restrict__ Ab, const __bf16* __restrict__ Wb,
                               const float* __restrict__ bias,
                               float* __restrict__ Cf, __bf16* __restrict__ Cb,
                               int Nrows, int applyLeaky) {
  __shared__ __bf16 at[TILE][H + 8];
  __shared__ __bf16 wt[H][H + 8];
  __shared__ float  bs[H];
  int tid = threadIdx.x;
  int rowBase = blockIdx.x * TILE;
  if (tid < H) bs[tid] = bias[tid];
  #pragma unroll
  for (int it = 0; it < 8; ++it) {             // W: 128 rows x 16 x 16B
    int i = tid + it * 256, r = i >> 4, c = i & 15;
    async_cp16((char*)&wt[r][0] + c * 16, (const char*)Wb + (size_t)i * 16);
  }
  #pragma unroll
  for (int it = 0; it < 4; ++it) {             // A: 64 rows x 16 x 16B (rows < NPAD)
    int i = tid + it * 256, r = i >> 4, c = i & 15;
    async_cp16((char*)&at[r][0] + c * 16,
               (const char*)Ab + (size_t)(rowBase + r) * (H * 2) + c * 16);
  }
  async_wait0();
  __syncthreads();
  int lane = tid & 31;
  int colTile = tid >> 5;
  int kbA = (lane & 16) ? 8 : 0;
  int m   = lane & 15;
  int kB  = (lane & 15) + ((lane & 16) ? 16 : 0);
  for (int rowTile = 0; rowTile < 4; ++rowTile) {
    v8f acc = {};
    #pragma unroll
    for (int kk = 0; kk < 4; ++kk) {
      v16bf a, b;
      #pragma unroll
      for (int i = 0; i < 16; ++i) {
        int k = ((i >> 3) << 4) + kbA + (i & 7);
        a[i] = at[rowTile * 16 + m][kk * 32 + k];
        b[i] = wt[kk * 32 + k][colTile * 16 + m];
      }
      acc = __builtin_amdgcn_wmma_f32_16x16x32_bf16(false, a, false, b,
                                                    (short)0, acc, false, false);
    }
    int mo  = (lane & 16) ? 8 : 0;
    int col = colTile * 16 + (lane & 15);
    #pragma unroll
    for (int j = 0; j < 8; ++j) {
      int row = rowBase + rowTile * 16 + j + mo;
      if (row < Nrows) {
        float v = acc[j] + bs[col];
        if (applyLeaky) v = lrelu(v);
        if (Cf) Cf[(size_t)row * H + col] = v;
        if (Cb) Cb[(size_t)row * H + col] = f2bf(v);
      }
    }
  }
}

// ---------------------------------------------------------------- fused RGCN message pass
// 64 sorted same-relation edges/block. Async-gather bf16 x[src] rows + W_rel into LDS,
// WMMA, scale C rows by per-edge norm, atomic-add f32 into out[dst] (L2-resident).
__global__ void edge_kernel(const __bf16* __restrict__ xb, float* __restrict__ out,
                            const int* __restrict__ ssrc_g, const int* __restrict__ sdst_g,
                            const int* __restrict__ srel_g, const float* __restrict__ snrm_g,
                            const __bf16* __restrict__ Wbf) {
  __shared__ __bf16 xt[TILE][H + 8];
  __shared__ __bf16 wt[H][H + 8];
  __shared__ int   ssrc[TILE];
  __shared__ int   sdst[TILE];
  __shared__ float snrm[TILE];
  int tid = threadIdx.x;
  long base = (long)blockIdx.x * TILE;
  if (tid < TILE) {
    ssrc[tid] = ssrc_g[base + tid];
    sdst[tid] = sdst_g[base + tid];
    snrm[tid] = snrm_g[base + tid];
  }
  __syncthreads();
  int rel = srel_g[base];                       // block-uniform by construction
  const __bf16* W = Wbf + (size_t)rel * H * H;
  #pragma unroll
  for (int it = 0; it < 8; ++it) {              // W_rel tile
    int i = tid + it * 256, r = i >> 4, c = i & 15;
    async_cp16((char*)&wt[r][0] + c * 16, (const char*)W + (size_t)i * 16);
  }
  #pragma unroll
  for (int it = 0; it < 4; ++it) {              // gathered source rows (pure byte copy)
    int i = tid + it * 256, r = i >> 4, c = i & 15;
    async_cp16((char*)&xt[r][0] + c * 16,
               (const char*)xb + (size_t)ssrc[r] * (H * 2) + c * 16);
  }
  async_wait0();
  __syncthreads();
  int lane = tid & 31;
  int colTile = tid >> 5;
  int kbA = (lane & 16) ? 8 : 0;
  int m   = lane & 15;
  int kB  = (lane & 15) + ((lane & 16) ? 16 : 0);
  for (int rowTile = 0; rowTile < 4; ++rowTile) {
    v8f acc = {};
    #pragma unroll
    for (int kk = 0; kk < 4; ++kk) {
      v16bf a, b;
      #pragma unroll
      for (int i = 0; i < 16; ++i) {
        int k = ((i >> 3) << 4) + kbA + (i & 7);
        a[i] = xt[rowTile * 16 + m][kk * 32 + k];
        b[i] = wt[kk * 32 + k][colTile * 16 + m];
      }
      acc = __builtin_amdgcn_wmma_f32_16x16x32_bf16(false, a, false, b,
                                                    (short)0, acc, false, false);
    }
    int col = colTile * 16 + (lane & 15);
    int mo  = (lane & 16) ? 8 : 0;
    #pragma unroll
    for (int j = 0; j < 8; ++j) {
      int mm = rowTile * 16 + j + mo;
      int d = sdst[mm];
      if (d >= 0) atomicAdd(&out[(size_t)d * H + col], acc[j] * snrm[mm]);
    }
  }
}

// ---------------------------------------------------------------- 128 -> 2 head
__global__ void out_kernel(const float* __restrict__ h, const float* __restrict__ W2,
                           const float* __restrict__ b2, float* __restrict__ out, int Nn) {
  int node = blockIdx.x;
  if (node >= Nn) return;
  int tid = threadIdx.x;          // 64
  int col = tid >> 5;
  int lane = tid & 31;
  float acc = 0.f;
  for (int k = lane; k < H; k += 32) acc += h[(long)node * H + k] * W2[k * 2 + col];
  for (int o = 16; o > 0; o >>= 1) acc += __shfl_down(acc, o, 32);
  if (lane == 0) out[node * 2 + col] = acc + b2[col];
}

// ---------------------------------------------------------------- launcher
extern "C" void kernel_launch(void* const* d_in, const int* in_sizes, int n_in,
                              void* d_out, int out_size, void* d_ws, size_t ws_size,
                              hipStream_t stream) {
  const float* feature = (const float*)d_in[0];
  const int*   eidx    = (const int*)d_in[1];
  const int*   etype   = (const int*)d_in[2];
  const float* W_des   = (const float*)d_in[3];
  const float* b_des   = (const float*)d_in[4];
  const float* W_tweet = (const float*)d_in[5];
  const float* b_tweet = (const float*)d_in[6];
  const float* W_num   = (const float*)d_in[7];
  const float* b_num   = (const float*)d_in[8];
  const float* W_cat   = (const float*)d_in[9];
  const float* b_cat   = (const float*)d_in[10];
  const float* W_in    = (const float*)d_in[11];
  const float* b_in    = (const float*)d_in[12];
  const float* rgcn_W  = (const float*)d_in[13];
  const float* rgcn_rt = (const float*)d_in[14];
  const float* rgcn_b  = (const float*)d_in[15];
  const float* W_out1  = (const float*)d_in[16];
  const float* b_out1  = (const float*)d_in[17];
  const float* W_out2  = (const float*)d_in[18];
  const float* b_out2  = (const float*)d_in[19];

  const int Nn = in_sizes[0] / FD;
  const int E  = in_sizes[2];
  const int* src = eidx;
  const int* dst = eidx + E;
  const int EPAD = ((E + 63) / 64) * 64 + RNUM * 64;
  const int gG   = (Nn + TILE - 1) / TILE;
  const int NPAD = gG * TILE;

  char* w = (char*)d_ws;
  size_t off = 0;
  auto alloc = [&](size_t bytes) {
    size_t o = off; off += (bytes + 255) & ~(size_t)255; return o;
  };
  float*  cnt    = (float*) (w + alloc((size_t)RNUM * Nn * 4));
  int*    hist   = (int*)   (w + alloc(RNUM * 4));
  int*    cursor = (int*)   (w + alloc(RNUM * 4));
  int*    ssrc   = (int*)   (w + alloc((size_t)EPAD * 4));
  int*    sdst   = (int*)   (w + alloc((size_t)EPAD * 4));
  int*    srel   = (int*)   (w + alloc((size_t)EPAD * 4));
  float*  snrm   = (float*) (w + alloc((size_t)EPAD * 4));
  float*  PA     = (float*) (w + alloc((size_t)Nn * H * 4));
  float*  PB     = (float*) (w + alloc((size_t)Nn * H * 4));
  __bf16* B0     = (__bf16*)(w + alloc((size_t)NPAD * H * 2));
  __bf16* B1     = (__bf16*)(w + alloc((size_t)NPAD * H * 2));
  __bf16* B2     = (__bf16*)(w + alloc((size_t)NPAD * H * 2));
  __bf16* WinB   = (__bf16*)(w + alloc((size_t)H * H * 2));
  __bf16* RtB    = (__bf16*)(w + alloc((size_t)H * H * 2));
  __bf16* Wo1B   = (__bf16*)(w + alloc((size_t)H * H * 2));
  __bf16* RWB    = (__bf16*)(w + alloc((size_t)RNUM * H * H * 2));

  hipMemsetAsync(cnt,  0,    (size_t)RNUM * Nn * 4, stream);
  hipMemsetAsync(hist, 0,    RNUM * 4,              stream);
  hipMemsetAsync(ssrc, 0,    (size_t)EPAD * 4,      stream);
  hipMemsetAsync(sdst, 0xFF, (size_t)EPAD * 4,      stream);   // dst = -1 sentinel
  hipMemsetAsync(srel, 0,    (size_t)EPAD * 4,      stream);
  hipMemsetAsync(snrm, 0,    (size_t)EPAD * 4,      stream);
  size_t tailB = (size_t)(NPAD - Nn) * H * 2;                  // zero mirror pad rows
  if (tailB) {
    hipMemsetAsync(B0 + (size_t)Nn * H, 0, tailB, stream);
    hipMemsetAsync(B1 + (size_t)Nn * H, 0, tailB, stream);
    hipMemsetAsync(B2 + (size_t)Nn * H, 0, tailB, stream);
  }

  // weights -> bf16 once
  fconv_kernel<<<(H * H + 255) / 256, 256, 0, stream>>>(W_in,    WinB, H * H);
  fconv_kernel<<<(H * H + 255) / 256, 256, 0, stream>>>(rgcn_rt, RtB,  H * H);
  fconv_kernel<<<(H * H + 255) / 256, 256, 0, stream>>>(W_out1,  Wo1B, H * H);
  fconv_kernel<<<(RNUM * H * H + 255) / 256, 256, 0, stream>>>(rgcn_W, RWB,
                                                               (long)RNUM * H * H);

  int gE = (E + 255) / 256;
  count_kernel  <<<gE, 256, 0, stream>>>(dst, etype, cnt, E, Nn);
  hist_kernel   <<<gE, 256, 0, stream>>>(etype, hist, E);
  scan_kernel   <<<1, 32, 0, stream>>>(hist, cursor);
  scatter_kernel<<<gE, 256, 0, stream>>>(src, dst, etype, cnt, cursor,
                                         ssrc, sdst, srel, snrm, E, Nn);

  // input embed + W_in
  stage1_kernel<<<(Nn + 15) / 16, 256, 0, stream>>>(feature, W_des, b_des, W_tweet,
                                                    b_tweet, W_num, b_num, W_cat, b_cat,
                                                    B0, Nn);
  gemm128_kernel<<<gG, 256, 0, stream>>>(B0, WinB, b_in, nullptr, B1, Nn, 1);

  int gEdge = EPAD / 64;
  int cg = (Nn * H + 255) / 256;
  // RGCN layer 1
  gemm128_kernel<<<gG, 256, 0, stream>>>(B1, RtB, rgcn_b, PA, nullptr, Nn, 0);
  edge_kernel   <<<gEdge, 256, 0, stream>>>(B1, PA, ssrc, sdst, srel, snrm, RWB);
  fconv_kernel  <<<cg, 256, 0, stream>>>(PA, B2, (long)Nn * H);
  // RGCN layer 2
  gemm128_kernel<<<gG, 256, 0, stream>>>(B2, RtB, rgcn_b, PB, nullptr, Nn, 0);
  edge_kernel   <<<gEdge, 256, 0, stream>>>(B2, PB, ssrc, sdst, srel, snrm, RWB);
  fconv_kernel  <<<cg, 256, 0, stream>>>(PB, B0, (long)Nn * H);
  // head
  gemm128_kernel<<<gG, 256, 0, stream>>>(B0, Wo1B, b_out1, PA, nullptr, Nn, 1);
  out_kernel    <<<Nn, 64, 0, stream>>>(PA, W_out2, b_out2, (float*)d_out, Nn);
}
